// BasicBlock_22308060135674
// MI455X (gfx1250) — matrix operand order repaired and checked by reading, not compiled, not run on hardware
//
#include <hip/hip_runtime.h>
#include <math.h>
#include <stdint.h>
#include <stddef.h>

// ---------------------------------------------------------------------------
// MI455X / gfx1250 implementation of the FISTA dict-conv BasicBlock.
// All convs run as implicit-GEMM on v_wmma_f32_16x16x32_bf16 (f32 accum).
// GEMM K dimension is ordered (kh, kw, c) with channel innermost so each
// 32-wide K-step has wave-uniform (kh,kw): no per-element divides/branches.
// Epilogue mode / output dtype selection is fully compile-time.
// ---------------------------------------------------------------------------

typedef __attribute__((ext_vector_type(16))) __bf16 v16bf;
typedef __attribute__((ext_vector_type(8)))  float  v8f;
typedef __attribute__((ext_vector_type(4)))  unsigned int v4u;

union FragBF { v16bf v; v4u q[2]; };

__device__ __forceinline__ uint16_t f2bf(float f) {
  union { float f; uint32_t u; } cv; cv.f = f;
  uint32_t u = cv.u;
  uint32_t r = u + 0x7FFFu + ((u >> 16) & 1u);   // round-to-nearest-even
  return (uint16_t)(r >> 16);
}

#define MU_STEP   0.1f
#define LMBD      0.1f
#define SHRINK_T  (MU_STEP * LMBD)
#define BN_EPS    1e-5f

// ---------------------------------------------------------------------------
// Weight prep: unit-L2-normalize each filter, emit bf16 in channel-innermost
// K order:
//   Afwd[co][(kh*ks+kw)*Cin  + ci]   (forward conv GEMM A)
//   Atr [ci][(kh*ks+kw)*Cout + co]   (conv-transpose GEMM A)
// ---------------------------------------------------------------------------
__global__ void prep_norm_weights(const float* __restrict__ W,
                                  uint16_t* __restrict__ Afwd,
                                  uint16_t* __restrict__ Atr,
                                  int Cout, int Cin, int ks) {
  const int co  = blockIdx.x;
  const int ks2 = ks * ks;
  const int K   = Cin * ks2;
  __shared__ float red[256];
  float s = 0.f;
  for (int i = threadIdx.x; i < K; i += 256) {
    float w = W[(size_t)co * K + i];
    s += w * w;
  }
  red[threadIdx.x] = s;
  __syncthreads();
  for (int st = 128; st > 0; st >>= 1) {
    if ((int)threadIdx.x < st) red[threadIdx.x] += red[threadIdx.x + st];
    __syncthreads();
  }
  const float inv  = 1.0f / (sqrtf(red[0]) + 1e-12f);
  const int   rowT = Cout * ks2;
  for (int i = threadIdx.x; i < K; i += 256) {
    float w = W[(size_t)co * K + i] * inv;   // OIHW: i = ci*ks2 + rr
    uint16_t b = f2bf(w);
    int ci = i / ks2;
    int rr = i - ci * ks2;
    Afwd[(size_t)co * K + rr * Cin + ci]     = b;
    Atr[(size_t)ci * rowT + rr * Cout + co]  = b;
  }
}

__global__ void convert_f32_bf16(const float* __restrict__ in,
                                 uint16_t* __restrict__ out, int n) {
  int i = blockIdx.x * 256 + threadIdx.x;
  if (i < n) out[i] = f2bf(in[i]);
}

__global__ void fill_zero_f32(float* __restrict__ p, int n) {
  int i = blockIdx.x * 256 + threadIdx.x;
  if (i < n) p[i] = 0.f;
}

// a = c + coef*(c - c_prev); emit f32 (aux for next conv) and bf16 (conv_t B)
__global__ void momentum_kernel(const float* __restrict__ c,
                                const float* __restrict__ cp, float coef,
                                float* __restrict__ aF,
                                uint16_t* __restrict__ aB, int n) {
  int i = blockIdx.x * 256 + threadIdx.x;
  if (i < n) {
    float cv = c[i];
    float v  = cv + coef * (cv - cp[i]);
    aF[i] = v;
    aB[i] = f2bf(v);
  }
}

// eval-mode BN; emit f32 and bf16 copies (bf16 feeds block-2 WMMA convs)
__global__ void bn_f32_bf16(const float* __restrict__ in,
                            const float* __restrict__ g, const float* __restrict__ b,
                            const float* __restrict__ m, const float* __restrict__ v,
                            float* __restrict__ outF, uint16_t* __restrict__ outB,
                            int C, int HW, int n) {
  int i = blockIdx.x * 256 + threadIdx.x;
  if (i >= n) return;
  int c = (i / HW) % C;
  float sc  = g[c] * rsqrtf(v[c] + BN_EPS);
  float val = (in[i] - m[c]) * sc + b[c];
  outF[i] = val;
  outB[i] = f2bf(val);
}

// out = relu( bn2(c2) + bnsc(sc) )
__global__ void bn_add_relu(const float* __restrict__ c2, const float* __restrict__ sc,
                            const float* __restrict__ g2, const float* __restrict__ b2,
                            const float* __restrict__ m2, const float* __restrict__ v2,
                            const float* __restrict__ gs, const float* __restrict__ bs,
                            const float* __restrict__ ms, const float* __restrict__ vs,
                            float* __restrict__ out, int C, int HW, int n) {
  int i = blockIdx.x * 256 + threadIdx.x;
  if (i >= n) return;
  int c = (i / HW) % C;
  float s2 = g2[c] * rsqrtf(v2[c] + BN_EPS);
  float ss = gs[c] * rsqrtf(vs[c] + BN_EPS);
  float v  = (c2[i] - m2[c]) * s2 + b2[c] + (sc[i] - ms[c]) * ss + bs[c];
  out[i] = fmaxf(v, 0.f);
}

// ---------------------------------------------------------------------------
// Implicit-GEMM conv / conv-transpose with WMMA bf16.
//   out[Mrow, p] = sum_k A[Mrow][k] * B[k][p]
//   K order: k = (kh*KS+kw)*Cact + c  (channel innermost; Cact % 32 == 0)
//   p decodes to (n, ho, wo).
//   TMODE==0: forward conv   h = ho*stride + kh - pad
//   TMODE==1: conv_transpose h = (ho + pad - kh)/stride (exact-div gather)
// EMODE (compile-time): 0 plain, 1 shrink(mu*acc), 2 shrink(aux + mu*acc),
//                       3 aux - acc.   WF/WB: emit f32 / bf16 output.
// Block tile 128x128, 8 waves (2M x 4N), wave tile 64x32 = 8 x v8f accum.
// ---------------------------------------------------------------------------
#define MT  128
#define NT  128
#define KT  32
#define LDA 40   // halves; padded to keep b128 alignment + reduce conflicts
#define LDB 40

template <int KS, int TMODE, int EMODE, bool WF, bool WB>
__global__ void __launch_bounds__(256)
conv_gemm_wmma(const uint16_t* __restrict__ A, const uint16_t* __restrict__ act,
               float* __restrict__ outF, uint16_t* __restrict__ outB,
               const float* __restrict__ aux,
               int Mtot, int Ktot, int Cact, int Hin, int Win,
               int Ho, int Wo, int stride, int pad) {
  __shared__ __align__(16) uint16_t lA[MT * LDA];
  __shared__ __align__(16) uint16_t lB[NT * LDB];

  const int tid  = threadIdx.x;
  const int lane = tid & 31;
  const int wid  = tid >> 5;
  const int wm   = wid >> 2;          // 0..1  (M half of block tile)
  const int wnq  = wid & 3;           // 0..3  (N quarter of block tile)
  const int hi   = (lane >= 16) ? 1 : 0;
  const int l15  = lane & 15;

  const int mBase = blockIdx.x * MT;
  const int pBase = blockIdx.y * NT;
  const int HoWo  = Ho * Wo;
  const int HinWin = Hin * Win;

  // This thread's B-gather column (fixed across the K loop)
  const int pl    = tid >> 1;         // local pixel 0..127
  const int khalf = (tid & 1) * 16;   // k sub-range 0 or 16
  const int p0    = pBase + pl;
  const int nb    = p0 / HoWo;
  const int hwrem = p0 - nb * HoWo;
  const int ho    = hwrem / Wo;
  const int wo    = hwrem - ho * Wo;
  const size_t actBatch = (size_t)nb * Cact * HinWin;

  const v8f vzero = {0.f, 0.f, 0.f, 0.f, 0.f, 0.f, 0.f, 0.f};
  v8f acc[4][2];
#pragma unroll
  for (int mi = 0; mi < 4; ++mi)
#pragma unroll
    for (int ni = 0; ni < 2; ++ni) acc[mi][ni] = vzero;

  int kh = 0, kw = 0;   // uniform filter tap for this K-step
  int c0 = 0;           // uniform channel base for this K-step
  for (int kB = 0; kB < Ktot; kB += KT) {
    __syncthreads();  // protect previous iteration's fragment reads

    {  // A tile: 128 rows x 32 halves, vectorized b128 global loads
      const int mloc = tid >> 1;
      const int koff = (tid & 1) * 16;
      const v4u* src = (const v4u*)(A + (size_t)(mBase + mloc) * Ktot + kB + koff);
      v4u* dst = (v4u*)&lA[mloc * LDA + koff];
      dst[0] = src[0];
      dst[1] = src[1];
      if (kB + KT < Ktot)
        __builtin_prefetch(A + (size_t)(mBase + mloc) * Ktot + kB + KT + koff, 0, 0);
    }
    {  // B tile: channel-strided gather; (kh,kw) uniform for whole K-step
      bool valid;
      int h, w;
      if (TMODE == 0) {
        h = ho * stride + kh - pad;
        w = wo * stride + kw - pad;
        valid = (h >= 0) & (h < Hin) & (w >= 0) & (w < Win);
      } else {
        const int hn = ho + pad - kh;
        const int wn = wo + pad - kw;
        h = hn / stride;
        w = wn / stride;
        valid = (hn >= 0) & (wn >= 0) & (hn % stride == 0) & (wn % stride == 0) &
                (h < Hin) & (w < Win);
      }
      uint32_t packed[8];
      if (valid) {
        const uint16_t* src =
            act + actBatch + (size_t)(c0 + khalf) * HinWin + h * Win + w;
#pragma unroll
        for (int ii = 0; ii < 8; ++ii) {
          uint32_t v0 = src[(2 * ii) * HinWin];
          uint32_t v1 = src[(2 * ii + 1) * HinWin];
          packed[ii] = v0 | (v1 << 16);
        }
      } else {
#pragma unroll
        for (int ii = 0; ii < 8; ++ii) packed[ii] = 0u;
      }
      v4u t0, t1;
      t0.x = packed[0]; t0.y = packed[1]; t0.z = packed[2]; t0.w = packed[3];
      t1.x = packed[4]; t1.y = packed[5]; t1.z = packed[6]; t1.w = packed[7];
      v4u* dstB = (v4u*)&lB[pl * LDB + khalf];
      dstB[0] = t0;
      dstB[1] = t1;
    }
    // advance uniform (kh, kw, c0) without any division
    c0 += KT;
    if (c0 >= Cact) {
      c0 = 0;
      if (KS > 1) {
        ++kw;
        if (kw == KS) { kw = 0; ++kh; }
      }
    }

    __syncthreads();

    // Fragment loads follow the CDNA5 bf16 WMMA VGPR layouts:
    // A lanes 0-15: K 0..7 / 16..23 ; lanes 16-31: K 8..15 / 24..31
    // B lanes 0-15: K 0..7 / 16..23 ; lanes 16-31: K 8..15 / 24..31
    FragBF fa[4], fb[2];
#pragma unroll
    for (int mi = 0; mi < 4; ++mi) {
      const int mrow = wm * 64 + mi * 16 + l15;
      const v4u* pa = (const v4u*)&lA[mrow * LDA + hi * 8];
      fa[mi].q[0] = pa[0];
      fa[mi].q[1] = pa[2];   // +16 halves
    }
#pragma unroll
    for (int ni = 0; ni < 2; ++ni) {
      const int prow = wnq * 32 + ni * 16 + l15;
      const v4u* pb = (const v4u*)&lB[prow * LDB + hi * 8];
      fb[ni].q[0] = pb[0];
      fb[ni].q[1] = pb[2];
    }
#pragma unroll
    for (int mi = 0; mi < 4; ++mi)
#pragma unroll
      for (int ni = 0; ni < 2; ++ni)
        acc[mi][ni] = __builtin_amdgcn_wmma_f32_16x16x32_bf16(
            false, fa[mi].v, false, fb[ni].v, (short)0, acc[mi][ni],
            false, false);
  }

  // Epilogue: C/D layout -> M = r + 8*hi (per frag), N = lane&15.
  // Straight-line: EMODE and output dtypes are compile-time.
#pragma unroll
  for (int ni = 0; ni < 2; ++ni) {
    const int pcol = pBase + wnq * 32 + ni * 16 + l15;
    const int nb2 = pcol / HoWo;
    const int hw2 = pcol - nb2 * HoWo;
    const int ho2 = hw2 / Wo;
    const int wo2 = hw2 - ho2 * Wo;
#pragma unroll
    for (int mi = 0; mi < 4; ++mi) {
      const int M0 = mBase + wm * 64 + mi * 16 + 8 * hi;
      const size_t base = ((size_t)(nb2 * Mtot + M0) * Ho + ho2) * Wo + wo2;
#pragma unroll
      for (int r = 0; r < 8; ++r) {
        const size_t oidx = base + (size_t)r * HoWo;
        float v = acc[mi][ni][r];
        if (EMODE == 1)      v = fmaxf(MU_STEP * v - SHRINK_T, 0.f);
        else if (EMODE == 2) v = fmaxf(aux[oidx] + MU_STEP * v - SHRINK_T, 0.f);
        else if (EMODE == 3) v = aux[oidx] - v;
        if (WF) outF[oidx] = v;
        if (WB) outB[oidx] = f2bf(v);
      }
    }
  }
}

// ---------------------------------------------------------------------------
// Host side
// ---------------------------------------------------------------------------
template <int KS, int TMODE, int EMODE, bool WF, bool WB>
static void launch_conv(const uint16_t* A, const uint16_t* act, float* outF,
                        uint16_t* outB, const float* aux,
                        int Mtot, int Ktot, int Cact, int Hin, int Win,
                        int Ho, int Wo, int stride, int pad, hipStream_t s) {
  const int Npix = 32 * Ho * Wo;          // N = 32 batch
  dim3 grid(Mtot / MT, Npix / NT);
  conv_gemm_wmma<KS, TMODE, EMODE, WF, WB><<<grid, 256, 0, s>>>(
      A, act, outF, outB, aux, Mtot, Ktot, Cact, Hin, Win, Ho, Wo, stride, pad);
}

static inline int ceil_div(int a, int b) { return (a + b - 1) / b; }

extern "C" void kernel_launch(void* const* d_in, const int* in_sizes, int n_in,
                              void* d_out, int out_size, void* d_ws, size_t ws_size,
                              hipStream_t stream) {
  (void)in_sizes; (void)n_in; (void)out_size; (void)ws_size;

  const float* x    = (const float*)d_in[0];   // (32,128,56,56)
  const float* W1   = (const float*)d_in[1];   // (256,128,3,3)
  const float* W2   = (const float*)d_in[2];   // (256,256,3,3)
  const float* Wsc  = (const float*)d_in[3];   // (256,128,1,1)
  const float* bn1g = (const float*)d_in[4];
  const float* bn1b = (const float*)d_in[5];
  const float* bn1m = (const float*)d_in[6];
  const float* bn1v = (const float*)d_in[7];
  const float* bn2g = (const float*)d_in[8];
  const float* bn2b = (const float*)d_in[9];
  const float* bn2m = (const float*)d_in[10];
  const float* bn2v = (const float*)d_in[11];
  const float* bscg = (const float*)d_in[12];
  const float* bscb = (const float*)d_in[13];
  const float* bscm = (const float*)d_in[14];
  const float* bscv = (const float*)d_in[15];
  float* out = (float*)d_out;

  const int nX = 32 * 128 * 56 * 56;   // 12,845,056
  const int nC = 32 * 256 * 28 * 28;   //  6,422,528

  // workspace arena
  char* ws = (char*)d_ws;
  size_t off = 0;
  auto arena = [&](size_t bytes) -> void* {
    void* p = ws + off;
    off = (off + bytes + 255) & ~(size_t)255;
    return p;
  };
  uint16_t* xB   = (uint16_t*)arena((size_t)nX * 2);
  uint16_t* A1   = (uint16_t*)arena((size_t)256 * 1152 * 2);
  uint16_t* A1t  = (uint16_t*)arena((size_t)128 * 2304 * 2);
  uint16_t* A2   = (uint16_t*)arena((size_t)256 * 2304 * 2);
  uint16_t* A2t  = (uint16_t*)arena((size_t)256 * 2304 * 2);
  uint16_t* Asc  = (uint16_t*)arena((size_t)256 * 128 * 2);
  float*    cA   = (float*)arena((size_t)nC * 4);
  float*    cB   = (float*)arena((size_t)nC * 4);
  float*    aF   = (float*)arena((size_t)nC * 4);
  uint16_t* aB   = (uint16_t*)arena((size_t)nC * 2);
  uint16_t* rB   = (uint16_t*)arena((size_t)nX * 2);  // max: block-1 residual
  float*    bn1F = (float*)arena((size_t)nC * 4);
  uint16_t* bn1B = (uint16_t*)arena((size_t)nC * 2);
  float*    scF  = (float*)arena((size_t)nC * 4);

  // --- prep: normalized bf16 weights, bf16 input ---
  prep_norm_weights<<<256, 256, 0, stream>>>(W1, A1, A1t, 256, 128, 3);
  prep_norm_weights<<<256, 256, 0, stream>>>(W2, A2, A2t, 256, 256, 3);
  convert_f32_bf16<<<ceil_div(256 * 128, 256), 256, 0, stream>>>(Wsc, Asc, 256 * 128);
  convert_f32_bf16<<<ceil_div(nX, 256), 256, 0, stream>>>(x, xB, nX);

  // ---------------- block 1: dict_conv(x, W1, stride=2) ----------------
  float* c_cur = cA;
  float* c_prev = cB;
  fill_zero_f32<<<ceil_div(nC, 256), 256, 0, stream>>>(c_prev, nC);
  // c = shrink(MU * conv(x, W, 2))
  launch_conv<3, 0, 1, true, false>(A1, xB, c_cur, nullptr, nullptr,
                                    256, 1152, 128, 56, 56, 28, 28, 2, 1, stream);
  double t = 1.0;
  for (int it = 0; it < 3; ++it) {
    const double tn = (1.0 + sqrt(1.0 + 4.0 * t * t)) / 2.0;
    const float coef = (float)((t - 1.0) / tn);
    t = tn;
    momentum_kernel<<<ceil_div(nC, 256), 256, 0, stream>>>(c_cur, c_prev, coef, aF, aB, nC);
    float* tmp = c_prev; c_prev = c_cur; c_cur = tmp;
    // r = x - conv_t(a, W, 2)   -> bf16 only (feeds next conv's B)
    launch_conv<3, 1, 3, false, true>(A1t, aB, nullptr, rB, x,
                                      128, 2304, 256, 28, 28, 56, 56, 2, 1, stream);
    // c = shrink(a + MU * conv(r, W, 2))
    launch_conv<3, 0, 2, true, false>(A1, rB, c_cur, nullptr, aF,
                                      256, 1152, 128, 56, 56, 28, 28, 2, 1, stream);
  }
  bn_f32_bf16<<<ceil_div(nC, 256), 256, 0, stream>>>(
      c_cur, bn1g, bn1b, bn1m, bn1v, bn1F, bn1B, 256, 28 * 28, nC);

  // ---------------- block 2: dict_conv(bn1, W2, stride=1) ----------------
  float* c2 = (c_cur == cA) ? cB : cA;   // free buffer
  float* c2p = c_cur;                    // overwrite old c (consumed by bn)
  fill_zero_f32<<<ceil_div(nC, 256), 256, 0, stream>>>(c2p, nC);
  launch_conv<3, 0, 1, true, false>(A2, bn1B, c2, nullptr, nullptr,
                                    256, 2304, 256, 28, 28, 28, 28, 1, 1, stream);
  t = 1.0;
  for (int it = 0; it < 3; ++it) {
    const double tn = (1.0 + sqrt(1.0 + 4.0 * t * t)) / 2.0;
    const float coef = (float)((t - 1.0) / tn);
    t = tn;
    momentum_kernel<<<ceil_div(nC, 256), 256, 0, stream>>>(c2, c2p, coef, aF, aB, nC);
    float* tmp = c2p; c2p = c2; c2 = tmp;
    launch_conv<3, 1, 3, false, true>(A2t, aB, nullptr, rB, bn1F,
                                      256, 2304, 256, 28, 28, 28, 28, 1, 1, stream);
    launch_conv<3, 0, 2, true, false>(A2, rB, c2, nullptr, aF,
                                      256, 2304, 256, 28, 28, 28, 28, 1, 1, stream);
  }

  // ---------------- shortcut: 1x1 stride-2 conv ----------------
  launch_conv<1, 0, 0, true, false>(Asc, xB, scF, nullptr, nullptr,
                                    256, 128, 128, 56, 56, 28, 28, 2, 0, stream);

  // ---------------- out = relu(bn2(c2) + bnsc(sc)) ----------------
  bn_add_relu<<<ceil_div(nC, 256), 256, 0, stream>>>(
      c2, scF, bn2g, bn2b, bn2m, bn2v, bscg, bscb, bscm, bscv,
      out, 256, 28 * 28, nC);
}
